// TransitionUpBlock_17841294147946
// MI455X (gfx1250) — matrix operand order repaired and checked by reading, not compiled, not run on hardware
//
#include <hip/hip_runtime.h>
#include <hip/hip_bf16.h>

typedef __attribute__((ext_vector_type(16))) __bf16 v16bf;
typedef __attribute__((ext_vector_type(8)))  float  v8f;

#define BATCH 16
#define N2    1024
#define N1    4096
#define C_OUT 256
#define K1    512
#define K2    256

// ---------------------------------------------------------------------------
// Pack weight matrix W (f32, Kd x 256, row-major) into bf16 WMMA B-fragment
// layout. Per 32x16 (KxN) tile (512 elems contiguous):
//   lane = ((klocal>>3)&1)*16 + nlocal, elem = (klocal&7) + (klocal>>4)*8   (CDNA5 16-bit B layout)
// ---------------------------------------------------------------------------
__global__ void pack_w_kernel(const float* __restrict__ W, __bf16* __restrict__ Wp,
                              int Kd) {
  int total = Kd * C_OUT;
  for (int idx = blockIdx.x * blockDim.x + threadIdx.x; idx < total;
       idx += gridDim.x * blockDim.x) {
    int tile = idx >> 9;     // 512 elems per tile
    int r    = idx & 511;
    int lane = r >> 4;
    int e    = r & 15;
    int kt   = tile >> 4;    // 16 N-tiles across 256 cols
    int nt   = tile & 15;
    int hi   = lane >> 4;
    int n    = nt * 16 + (lane & 15);
    int k    = kt * 32 + hi * 8 + (e & 7) + (e >> 3) * 16;
    Wp[idx] = (__bf16)W[(size_t)k * C_OUT + n];
  }
}

// ---------------------------------------------------------------------------
// Y[R x 256] = X[R x Kd] @ W[Kd x 256] + bias, via v_wmma_f32_16x16x32_bf16.
// Block = 128 threads = 4 waves; block owns 16 rows, wave w owns cols [64w,64w+64).
// A fragment (16x32 bf16) built on the fly from f32 activations:
//   lanes 0-15 : row = base+lane, elems 0-7 = K 0-7,  elems 8-15 = K 16-23
//   lanes 16-31: row = base+lane-16, elems 0-7 = K 8-15, elems 8-15 = K 24-31
// ---------------------------------------------------------------------------
__global__ __launch_bounds__(128)
void gemm_bias_kernel(const float* __restrict__ X, const __bf16* __restrict__ Wp,
                      const float* __restrict__ bias, float* __restrict__ Y, int Kd) {
  const int lane   = threadIdx.x & 31;
  const int wv     = threadIdx.x >> 5;     // 0..3
  const int laneLo = lane & 15;
  const int hi     = lane >> 4;
  const int rowBase = blockIdx.x * 16;

  v8f acc[4];
#pragma unroll
  for (int t = 0; t < 4; ++t) {
    v8f z = {0.f, 0.f, 0.f, 0.f, 0.f, 0.f, 0.f, 0.f};
    acc[t] = z;
  }

  const int ktiles = Kd >> 5;
  const float* xrowBase = X + (size_t)(rowBase + laneLo) * Kd + hi * 8;

  for (int kt = 0; kt < ktiles; ++kt) {
    const float* xr = xrowBase + kt * 32;
    float4 f0 = *(const float4*)(xr + 0);
    float4 f1 = *(const float4*)(xr + 4);
    float4 f2 = *(const float4*)(xr + 16);
    float4 f3 = *(const float4*)(xr + 20);
    v16bf a;
    a[0]  = (__bf16)f0.x; a[1]  = (__bf16)f0.y; a[2]  = (__bf16)f0.z; a[3]  = (__bf16)f0.w;
    a[4]  = (__bf16)f1.x; a[5]  = (__bf16)f1.y; a[6]  = (__bf16)f1.z; a[7]  = (__bf16)f1.w;
    a[8]  = (__bf16)f2.x; a[9]  = (__bf16)f2.y; a[10] = (__bf16)f2.z; a[11] = (__bf16)f2.w;
    a[12] = (__bf16)f3.x; a[13] = (__bf16)f3.y; a[14] = (__bf16)f3.z; a[15] = (__bf16)f3.w;

    const __bf16* wb = Wp + ((size_t)kt * 16 + wv * 4) * 512 + lane * 16;
    v16bf b0 = *(const v16bf*)(wb);
    v16bf b1 = *(const v16bf*)(wb + 512);
    v16bf b2 = *(const v16bf*)(wb + 1024);
    v16bf b3 = *(const v16bf*)(wb + 1536);

    acc[0] = __builtin_amdgcn_wmma_f32_16x16x32_bf16(false, a, false, b0, (short)0, acc[0], false, false);
    acc[1] = __builtin_amdgcn_wmma_f32_16x16x32_bf16(false, a, false, b1, (short)0, acc[1], false, false);
    acc[2] = __builtin_amdgcn_wmma_f32_16x16x32_bf16(false, a, false, b2, (short)0, acc[2], false, false);
    acc[3] = __builtin_amdgcn_wmma_f32_16x16x32_bf16(false, a, false, b3, (short)0, acc[3], false, false);
  }

  // D layout: lanes 0-15 -> M = e, N = lane ; lanes 16-31 -> M = 8+e, N = lane-16
  const int rB = rowBase + hi * 8;
#pragma unroll
  for (int t = 0; t < 4; ++t) {
    int col = wv * 64 + t * 16 + laneLo;
    float bc = bias[col];
#pragma unroll
    for (int e = 0; e < 8; ++e) {
      Y[(size_t)(rB + e) * C_OUT + col] = acc[t][e] + bc;
    }
  }
}

// ---------------------------------------------------------------------------
// Stage-1 column stats: per-block partial sum / sum-of-squares per channel.
// R must be divisible by gridDim.x. Deterministic (no atomics).
// ---------------------------------------------------------------------------
__global__ __launch_bounds__(256)
void colstats_kernel(const float* __restrict__ Y, int R, float* __restrict__ P) {
  int c   = threadIdx.x;
  int per = R / gridDim.x;
  int r0  = blockIdx.x * per;
  float s = 0.f, q = 0.f;
  for (int r = 0; r < per; ++r) {
    float v = Y[(size_t)(r0 + r) * C_OUT + c];
    s += v;
    q = fmaf(v, v, q);
  }
  P[(size_t)blockIdx.x * 512 + c]       = s;
  P[(size_t)blockIdx.x * 512 + 256 + c] = q;
}

// ---------------------------------------------------------------------------
// Stage-2: fold partials -> per-channel BN affine coefs (y*scale + shift).
// coef layout: [scale1 | shift1 | scale2 | shift2], each 256 floats.
// ---------------------------------------------------------------------------
__global__ __launch_bounds__(256)
void bn_finalize_kernel(const float* __restrict__ P1, int nb1, int R1,
                        const float* __restrict__ P2, int nb2, int R2,
                        const float* __restrict__ gamma, const float* __restrict__ beta,
                        float* __restrict__ coef) {
  int c = threadIdx.x;
  float g = gamma[c], be = beta[c];

  float s = 0.f, q = 0.f;
  for (int b = 0; b < nb1; ++b) { s += P1[b * 512 + c]; q += P1[b * 512 + 256 + c]; }
  float mu  = s / (float)R1;
  float var = q / (float)R1 - mu * mu;
  float inv = rsqrtf(var + 1e-5f);
  coef[c]       = g * inv;
  coef[256 + c] = be - mu * g * inv;

  s = 0.f; q = 0.f;
  for (int b = 0; b < nb2; ++b) { s += P2[b * 512 + c]; q += P2[b * 512 + 256 + c]; }
  mu  = s / (float)R2;
  var = q / (float)R2 - mu * mu;
  inv = rsqrtf(var + 1e-5f);
  coef[512 + c] = g * inv;
  coef[768 + c] = be - mu * g * inv;
}

// ---------------------------------------------------------------------------
// Fused kNN(3) inverse-distance interpolation + BN/ReLU of both branches + add.
// Block = 256 threads, handles 256 high-res points of one batch.
// Phase 1: thread = point, scans 1024 LDS-cached candidates for top-3.
// Phase 2: thread = channel, loops the 256 points (broadcast idx/w from LDS).
// ---------------------------------------------------------------------------
__global__ __launch_bounds__(256)
void interp_kernel(const float* __restrict__ pxyz, const float* __restrict__ sxyz,
                   const float* __restrict__ Y1, const float* __restrict__ Y2,
                   const float* __restrict__ coef, float* __restrict__ outF) {
  __shared__ float sx[N2], sy[N2], sz[N2];
  __shared__ int   nidx[256 * 3];
  __shared__ float nwgt[256 * 3];

  const int b  = blockIdx.y;
  const int n0 = blockIdx.x * 256;
  const int t  = threadIdx.x;

  for (int j = t; j < N2; j += 256) {
    const float* p = pxyz + ((size_t)b * N2 + j) * 3;
    sx[j] = p[0]; sy[j] = p[1]; sz[j] = p[2];
  }
  __syncthreads();

  {
    const float* qp = sxyz + ((size_t)b * N1 + n0 + t) * 3;
    float qx = qp[0], qy = qp[1], qz = qp[2];
    float d0 = 3.4e38f, d1 = 3.4e38f, d2 = 3.4e38f;
    int   i0 = 0, i1 = 0, i2 = 0;
    for (int j = 0; j < N2; ++j) {
      float dx = qx - sx[j], dy = qy - sy[j], dz = qz - sz[j];
      float d  = fmaf(dx, dx, fmaf(dy, dy, dz * dz));
      if (d < d2) {
        if (d < d1) {
          d2 = d1; i2 = i1;
          if (d < d0) { d1 = d0; i1 = i0; d0 = d; i0 = j; }
          else        { d1 = d;  i1 = j; }
        } else { d2 = d; i2 = j; }
      }
    }
    float w0 = 1.f / (sqrtf(d0) + 1e-8f);
    float w1 = 1.f / (sqrtf(d1) + 1e-8f);
    float w2 = 1.f / (sqrtf(d2) + 1e-8f);
    float wi = 1.f / (w0 + w1 + w2);
    nidx[t * 3 + 0] = i0; nidx[t * 3 + 1] = i1; nidx[t * 3 + 2] = i2;
    nwgt[t * 3 + 0] = w0 * wi; nwgt[t * 3 + 1] = w1 * wi; nwgt[t * 3 + 2] = w2 * wi;
  }
  __syncthreads();

  const float sc1 = coef[t], sh1 = coef[256 + t];
  const float sc2 = coef[512 + t], sh2 = coef[768 + t];
  const float* Y1b = Y1 + (size_t)b * N2 * C_OUT;

  for (int p = 0; p < 256; ++p) {
    int   ia = nidx[p * 3 + 0], ib = nidx[p * 3 + 1], ic = nidx[p * 3 + 2];
    float wa = nwgt[p * 3 + 0], wb = nwgt[p * 3 + 1], wc = nwgt[p * 3 + 2];
    float ga = fmaxf(fmaf(Y1b[(size_t)ia * C_OUT + t], sc1, sh1), 0.f);
    float gb = fmaxf(fmaf(Y1b[(size_t)ib * C_OUT + t], sc1, sh1), 0.f);
    float gc = fmaxf(fmaf(Y1b[(size_t)ic * C_OUT + t], sc1, sh1), 0.f);
    size_t ro = ((size_t)b * N1 + n0 + p) * C_OUT + t;
    float  sk = fmaxf(fmaf(Y2[ro], sc2, sh2), 0.f);
    outF[ro] = fmaf(wa, ga, fmaf(wb, gb, fmaf(wc, gc, sk)));
  }
}

// ---------------------------------------------------------------------------
extern "C" void kernel_launch(void* const* d_in, const int* in_sizes, int n_in,
                              void* d_out, int out_size, void* d_ws, size_t ws_size,
                              hipStream_t stream) {
  const float* points_xyz       = (const float*)d_in[0];
  const float* points_features  = (const float*)d_in[1];
  const float* skipped_xyz      = (const float*)d_in[2];
  const float* skipped_features = (const float*)d_in[3];
  const float* W1a   = (const float*)d_in[4];
  const float* b1a   = (const float*)d_in[5];
  const float* W1b   = (const float*)d_in[6];
  const float* b1b   = (const float*)d_in[7];
  const float* gamma = (const float*)d_in[8];
  const float* beta  = (const float*)d_in[9];
  float* out = (float*)d_out;

  // Workspace layout (~85.1 MB total)
  char* ws = (char*)d_ws;
  float*  Y1   = (float*)(ws + 0);                 // 16384*256*4  = 16,777,216 B
  float*  Y2   = (float*)(ws + 16777216);          // 65536*256*4  = 67,108,864 B
  __bf16* Wp1  = (__bf16*)(ws + 83886080);         // 512*256*2    =    262,144 B
  __bf16* Wp2  = (__bf16*)(ws + 84148224);         // 256*256*2    =    131,072 B
  float*  P1   = (float*)(ws + 84279296);          // 128*512*4    =    262,144 B
  float*  P2   = (float*)(ws + 84541440);          // 256*512*4    =    524,288 B
  float*  coef = (float*)(ws + 85065728);          // 4*256*4      =      4,096 B

  // Output part 1: skipped_xyz passthrough
  hipMemcpyAsync(out, skipped_xyz, (size_t)BATCH * N1 * 3 * sizeof(float),
                 hipMemcpyDeviceToDevice, stream);

  // Pack weights into WMMA B-fragment layout (bf16)
  pack_w_kernel<<<256, 256, 0, stream>>>(W1a, Wp1, K1);
  pack_w_kernel<<<128, 256, 0, stream>>>(W1b, Wp2, K2);

  // GEMMs (bf16 WMMA, f32 accumulate) + bias -> workspace
  gemm_bias_kernel<<<(BATCH * N2) / 16, 128, 0, stream>>>(points_features, Wp1, b1a, Y1, K1);
  gemm_bias_kernel<<<(BATCH * N1) / 16, 128, 0, stream>>>(skipped_features, Wp2, b1b, Y2, K2);

  // BatchNorm statistics (deterministic two-stage reduction)
  colstats_kernel<<<128, 256, 0, stream>>>(Y1, BATCH * N2, P1);
  colstats_kernel<<<256, 256, 0, stream>>>(Y2, BATCH * N1, P2);
  bn_finalize_kernel<<<1, 256, 0, stream>>>(P1, 128, BATCH * N2,
                                            P2, 256, BATCH * N1, gamma, beta, coef);

  // Fused kNN interpolation + BN/ReLU + skip add
  dim3 grid(N1 / 256, BATCH);
  interp_kernel<<<grid, 256, 0, stream>>>(points_xyz, skipped_xyz, Y1, Y2, coef,
                                          out + (size_t)BATCH * N1 * 3);
}
